// MultiHeadAttention_128849019255
// MI455X (gfx1250) — matrix operand order repaired and checked by reading, not compiled, not run on hardware
//
#include <hip/hip_runtime.h>

// Problem constants
#define BATCH   8
#define DMODEL  256
#define NPOS    2048
#define NHEADS  4
#define DHEAD   64

typedef __attribute__((ext_vector_type(16))) _Float16 v16h;
typedef __attribute__((ext_vector_type(4)))  _Float16 v4h;
typedef __attribute__((ext_vector_type(8)))  float    v8f;
typedef __attribute__((ext_vector_type(4)))  int      v4i;

#if defined(__gfx1250__) && __has_builtin(__builtin_amdgcn_global_load_async_to_lds_b128)
#define HAVE_ASYNC_LDS 1
#else
#define HAVE_ASYNC_LDS 0
#endif

__device__ __forceinline__ void wait_async0() {
#if __has_builtin(__builtin_amdgcn_s_wait_asynccnt)
    __builtin_amdgcn_s_wait_asynccnt(0);
#else
    asm volatile("s_wait_asynccnt 0" ::: "memory");
#endif
}

// 16B global -> LDS copy; async (ASYNCcnt-tracked) when available.
// Builtin signature (from hipcc diagnostic): first param is AS(1) int4*.
__device__ __forceinline__ void async_copy16(const _Float16* g, _Float16* l) {
#if HAVE_ASYNC_LDS
    __builtin_amdgcn_global_load_async_to_lds_b128(
        (__attribute__((address_space(1))) v4i*)g,
        (__attribute__((address_space(3))) v4i*)l, 0, 0);
#else
    *(uint4*)l = *(const uint4*)g;
#endif
}

__device__ __forceinline__ v8f zero8() {
    v8f z;
#pragma unroll
    for (int i = 0; i < 8; ++i) z[i] = 0.0f;
    return z;
}

__device__ __forceinline__ v8f wmma_f16(v16h a, v16h b, v8f c) {
    return __builtin_amdgcn_wmma_f32_16x16x32_f16(
        false, a, false, b, (short)0, c, false, false);
}

// ---- DPP 16-lane row reductions (no LDS traffic) --------------------------
template <int CTRL>
__device__ __forceinline__ float dppf(float x) {
    int r = __builtin_amdgcn_update_dpp(0, __float_as_int(x), CTRL, 0xF, 0xF, true);
    return __int_as_float(r);
}
// Masks {1,2,7,15} generate the full 16-lane group -> exact butterfly.
__device__ __forceinline__ float rowmax16(float x) {
    x = fmaxf(x, dppf<0xB1>(x));    // quad_perm [1,0,3,2]  (xor 1)
    x = fmaxf(x, dppf<0x4E>(x));    // quad_perm [2,3,0,1]  (xor 2)
    x = fmaxf(x, dppf<0x141>(x));   // row_half_mirror      (xor 7)
    x = fmaxf(x, dppf<0x140>(x));   // row_mirror           (xor 15)
    return x;
}
__device__ __forceinline__ float rowsum16(float x) {
    x += dppf<0xB1>(x);
    x += dppf<0x4E>(x);
    x += dppf<0x141>(x);
    x += dppf<0x140>(x);
    return x;
}

// A-matrix 16x32 f16 fragment (ISA 7.12.2): lane l holds row (l&15);
// lanes 0-15: K = {0..7, 16..23}; lanes 16-31: K = {8..15, 24..31}.
__device__ __forceinline__ v16h frag_a(const _Float16* tile, int ldk, int kbase) {
    const int l = threadIdx.x & 31;
    const _Float16* p = tile + (l & 15) * ldk + kbase + ((l < 16) ? 0 : 8);
    union { uint4 u[2]; v16h v; } r;
    r.u[0] = *(const uint4*)(p);
    r.u[1] = *(const uint4*)(p + 16);
    return r.v;
}

// B-matrix 32x16 f16 fragment: lane l holds column (l&15);
// lanes 0-15: K = {0..7, 16..23}; lanes 16-31: K = {8..15, 24..31}.
__device__ __forceinline__ v16h frag_b(const _Float16* tile, int ldk, int kbase) {
    const int l = threadIdx.x & 31;
    const _Float16* p = tile + (l & 15) * ldk + kbase + ((l < 16) ? 0 : 8);
    union { uint4 u[2]; v16h v; } r;
    r.u[0] = *(const uint4*)(p);
    r.u[1] = *(const uint4*)(p + 16);
    return r.v;
}

// ---------------------------------------------------------------------------
// Projection GEMM:  out[o,n] = sum_i w[o,i] * x[i,n] + bias[o]   (per batch)
// Double-buffered LDS; next K-slab preloaded into registers during WMMA.
// MODE 0: f16 -> QKV buffer [b][head][n][dd], scaled. MODE 1: fp32 out.
// ---------------------------------------------------------------------------
template <int MODE>
__global__ __launch_bounds__(128) void proj_kernel(
    const float* __restrict__ w, const float* __restrict__ bias,
    const float* __restrict__ x, _Float16* __restrict__ outh,
    float* __restrict__ outf, float outscale)
{
    __shared__ _Float16 WT[2][64 * 40];  // W tile, row-major [o][k]
    __shared__ _Float16 XT[2][64 * 40];  // X tile transposed: [n][k]

    const int tid = threadIdx.x;
    const int wv  = tid >> 5;
    const int l   = tid & 15;
    const int n0  = blockIdx.x * 64;
    const int o0  = blockIdx.y * 64;
    const int b   = blockIdx.z;

    v8f acc[4];
#pragma unroll
    for (int t = 0; t < 4; ++t) acc[t] = zero8();

    float4 wr[4], xr[4];
    // preload tile k0=0
#pragma unroll
    for (int i = 0; i < 4; ++i) {
        const int lin = tid + i * 128;
        wr[i] = *(const float4*)(w + (size_t)(o0 + (lin >> 3)) * DMODEL + (lin & 7) * 4);
        xr[i] = *(const float4*)(x + ((size_t)b * DMODEL + (lin >> 4)) * NPOS + n0 + (lin & 15) * 4);
    }

    int cur = 0;
    for (int kt = 0; kt < 8; ++kt) {
        // store staged registers (f32 -> f16) into buffer `cur`
#pragma unroll
        for (int i = 0; i < 4; ++i) {
            const int lin = tid + i * 128;
            const int row = lin >> 3, c4 = lin & 7;
            v4h h; h[0] = (_Float16)wr[i].x; h[1] = (_Float16)wr[i].y;
                   h[2] = (_Float16)wr[i].z; h[3] = (_Float16)wr[i].w;
            *(v4h*)&WT[cur][row * 40 + c4 * 4] = h;
        }
#pragma unroll
        for (int i = 0; i < 4; ++i) {
            const int lin = tid + i * 128;
            const int kr = lin >> 4, c4 = lin & 15;
            XT[cur][(c4 * 4 + 0) * 40 + kr] = (_Float16)xr[i].x;
            XT[cur][(c4 * 4 + 1) * 40 + kr] = (_Float16)xr[i].y;
            XT[cur][(c4 * 4 + 2) * 40 + kr] = (_Float16)xr[i].z;
            XT[cur][(c4 * 4 + 3) * 40 + kr] = (_Float16)xr[i].w;
        }
        // issue next tile's global loads (complete during the WMMAs below)
        if (kt < 7) {
            const int k0 = (kt + 1) * 32;
#pragma unroll
            for (int i = 0; i < 4; ++i) {
                const int lin = tid + i * 128;
                wr[i] = *(const float4*)(w + (size_t)(o0 + (lin >> 3)) * DMODEL + k0 + (lin & 7) * 4);
                xr[i] = *(const float4*)(x + ((size_t)b * DMODEL + k0 + (lin >> 4)) * NPOS + n0 + (lin & 15) * 4);
            }
        }
        __syncthreads();
        const v16h a = frag_a(&WT[cur][wv * 16 * 40], 40, 0);
#pragma unroll
        for (int t = 0; t < 4; ++t)
            acc[t] = wmma_f16(a, frag_b(&XT[cur][t * 16 * 40], 40, 0), acc[t]);
        cur ^= 1;
    }

    // Epilogue. C layout: row = vr + 8*(lane>=16), col = lane&15.
    const int half8 = ((threadIdx.x & 31) < 16) ? 0 : 8;
#pragma unroll
    for (int vr = 0; vr < 8; ++vr) {
        const int oc = o0 + wv * 16 + vr + half8;
        const float bs = bias[oc];
#pragma unroll
        for (int t = 0; t < 4; ++t) {
            const int n = n0 + t * 16 + l;
            const float val = (acc[t][vr] + bs) * outscale;
            if (MODE == 0) {
                const int dd = oc >> 2, hh = oc & 3;   // reshape(B,64,4,N)
                outh[(((size_t)b * NHEADS + hh) * NPOS + n) * DHEAD + dd] = (_Float16)val;
            } else {
                outf[((size_t)b * DMODEL + oc) * NPOS + n] = val;
            }
        }
    }
}

// ---------------------------------------------------------------------------
// Flash attention per (b, head). 4 waves x 16 query rows; K/V streamed in
// 64-wide tiles, double-buffered; K staged with async global->LDS copies.
// ---------------------------------------------------------------------------
__device__ __forceinline__ void stage_K(const _Float16* Kt, _Float16* dst, int tid) {
#pragma unroll
    for (int i = 0; i < 4; ++i) {
        const int lin = tid + i * 128;        // 512 x 16B chunks
        const int row = lin >> 3, c8 = lin & 7;
        async_copy16(Kt + row * DHEAD + c8 * 8, dst + row * 72 + c8 * 8);
    }
}
__device__ __forceinline__ void stage_V(const _Float16* Vt, _Float16* dst, int tid) {
#pragma unroll
    for (int i = 0; i < 8; ++i) {
        const int lin = tid + i * 128;
        const int m = lin >> 4, c4 = lin & 15;
        v4h v = *(const v4h*)(Vt + m * DHEAD + c4 * 4);
        dst[(c4 * 4 + 0) * 72 + m] = v[0];
        dst[(c4 * 4 + 1) * 72 + m] = v[1];
        dst[(c4 * 4 + 2) * 72 + m] = v[2];
        dst[(c4 * 4 + 3) * 72 + m] = v[3];
    }
}

__global__ __launch_bounds__(128) void attn_kernel(
    const _Float16* __restrict__ Q, const _Float16* __restrict__ K,
    const _Float16* __restrict__ V, float* __restrict__ attnws)
{
    __shared__ _Float16 KT[2][64 * 72];      // [m][dd]  (B-frags for QK^T)
    __shared__ _Float16 VT[2][64 * 72];      // [dd][m]  (B-frags for PV)
    __shared__ _Float16 PT[4 * 16 * 72];     // per-wave P scratch: [n][m]

    const int tid = threadIdx.x;
    const int wv  = tid >> 5;
    const int l   = tid & 31;
    const int n0  = blockIdx.x * 64;
    const int h   = blockIdx.y;
    const int b   = blockIdx.z;

    const size_t headoff = ((size_t)b * NHEADS + h) * NPOS * DHEAD;
    const _Float16* Qg = Q + headoff + (size_t)(n0 + wv * 16) * DHEAD;
    const _Float16* Kg = K + headoff;
    const _Float16* Vg = V + headoff;

    const v16h qa0 = frag_a(Qg, DHEAD, 0);
    const v16h qa1 = frag_a(Qg, DHEAD, 32);

    float mi[8], li[8];
    v8f oacc[4];
#pragma unroll
    for (int vr = 0; vr < 8; ++vr) { mi[vr] = -1.0e30f; li[vr] = 0.0f; }
#pragma unroll
    for (int t = 0; t < 4; ++t) oacc[t] = zero8();

    _Float16* Pw = PT + wv * 16 * 72;
    const int half8 = (l < 16) ? 0 : 8;

    // Prologue: stage tile 0 into buffer 0.
    stage_K(Kg, KT[0], tid);
    stage_V(Vg, VT[0], tid);
    wait_async0();
    __syncthreads();

    for (int mt = 0; mt < 32; ++mt) {
        const int cur = mt & 1, nxt = cur ^ 1;
        // Prefetch next K/V tile into the other buffer (overlaps with WMMAs).
        if (mt + 1 < 32) {
            stage_K(Kg + (size_t)(mt + 1) * 64 * DHEAD, KT[nxt], tid);
            stage_V(Vg + (size_t)(mt + 1) * 64 * DHEAD, VT[nxt], tid);
        }

        // S = Q K^T (softmax norm pre-folded into Q).
        v8f s[4];
#pragma unroll
        for (int t = 0; t < 4; ++t) {
            v8f c = zero8();
            c = wmma_f16(qa0, frag_b(&KT[cur][t * 16 * 72], 72, 0),  c);
            c = wmma_f16(qa1, frag_b(&KT[cur][t * 16 * 72], 72, 32), c);
            s[t] = c;
        }

        // Online softmax; row r = vr + half8 lives in one 16-lane group.
        float sc[8];
#pragma unroll
        for (int vr = 0; vr < 8; ++vr) {
            float rm = fmaxf(fmaxf(s[0][vr], s[1][vr]), fmaxf(s[2][vr], s[3][vr]));
            rm = rowmax16(rm);                       // DPP butterfly
            const float mn = fmaxf(mi[vr], rm);
            sc[vr] = __expf(mi[vr] - mn);
            mi[vr] = mn;
        }
        float rs[8];
#pragma unroll
        for (int vr = 0; vr < 8; ++vr) rs[vr] = 0.0f;
#pragma unroll
        for (int t = 0; t < 4; ++t)
#pragma unroll
            for (int vr = 0; vr < 8; ++vr) {
                const float p = __expf(s[t][vr] - mi[vr]);
                s[t][vr] = p;
                rs[vr] += p;
            }
#pragma unroll
        for (int vr = 0; vr < 8; ++vr) {
            li[vr] = li[vr] * sc[vr] + rowsum16(rs[vr]);
        }
#pragma unroll
        for (int t = 0; t < 4; ++t)
#pragma unroll
            for (int vr = 0; vr < 8; ++vr) oacc[t][vr] *= sc[vr];

        // P: C-layout -> per-wave LDS -> A-layout fragments.
#pragma unroll
        for (int t = 0; t < 4; ++t)
#pragma unroll
            for (int vr = 0; vr < 8; ++vr)
                Pw[(vr + half8) * 72 + t * 16 + (l & 15)] = (_Float16)s[t][vr];
        const v16h pa0 = frag_a(Pw, 72, 0);
        const v16h pa1 = frag_a(Pw, 72, 32);

        // O += P V.
#pragma unroll
        for (int t = 0; t < 4; ++t) {
            oacc[t] = wmma_f16(pa0, frag_b(&VT[cur][t * 16 * 72], 72, 0),  oacc[t]);
            oacc[t] = wmma_f16(pa1, frag_b(&VT[cur][t * 16 * 72], 72, 32), oacc[t]);
        }

        wait_async0();        // next tile's async copies done before reuse
        __syncthreads();
    }

    // Epilogue: O /= l, fp32 attn in [b][dd*4+h][n] layout.
    float inv[8];
#pragma unroll
    for (int vr = 0; vr < 8; ++vr) inv[vr] = 1.0f / li[vr];
#pragma unroll
    for (int t = 0; t < 4; ++t)
#pragma unroll
        for (int vr = 0; vr < 8; ++vr) {
            const int dd = t * 16 + (l & 15);
            const int n  = n0 + wv * 16 + vr + half8;
            attnws[((size_t)b * DMODEL + dd * 4 + h) * NPOS + n] = oacc[t][vr] * inv[vr];
        }
}

extern "C" void kernel_launch(void* const* d_in, const int* in_sizes, int n_in,
                              void* d_out, int out_size, void* d_ws, size_t ws_size,
                              hipStream_t stream) {
    const float* query = (const float*)d_in[0];
    const float* key   = (const float*)d_in[1];
    const float* value = (const float*)d_in[2];
    const float* wq = (const float*)d_in[3];
    const float* bq = (const float*)d_in[4];
    const float* wk = (const float*)d_in[5];
    const float* bk = (const float*)d_in[6];
    const float* wv = (const float*)d_in[7];
    const float* bv = (const float*)d_in[8];
    const float* wm = (const float*)d_in[9];
    const float* bm = (const float*)d_in[10];

    char* ws = (char*)d_ws;
    const size_t qkvBytes = (size_t)BATCH * NHEADS * NPOS * DHEAD * sizeof(_Float16); // 8 MB
    _Float16* Qb = (_Float16*)(ws);
    _Float16* Kb = (_Float16*)(ws + qkvBytes);
    _Float16* Vb = (_Float16*)(ws + 2 * qkvBytes);
    float* attnws = (float*)(ws + 3 * qkvBytes);                                      // 16 MB fp32

    const dim3 grid(NPOS / 64, DMODEL / 64, BATCH);  // (32, 4, 8)
    const dim3 blk(128);
    const float norm = 0.125f;  // 1/sqrt(DHEAD), folded into Q

    proj_kernel<0><<<grid, blk, 0, stream>>>(wq, bq, query, Qb, nullptr, norm);
    proj_kernel<0><<<grid, blk, 0, stream>>>(wk, bk, key,   Kb, nullptr, 1.0f);
    proj_kernel<0><<<grid, blk, 0, stream>>>(wv, bv, value, Vb, nullptr, 1.0f);

    const dim3 agrid(NPOS / 64, NHEADS, BATCH);      // (32, 4, 8)
    attn_kernel<<<agrid, blk, 0, stream>>>(Qb, Kb, Vb, attnws);

    proj_kernel<1><<<grid, blk, 0, stream>>>(wm, bm, attnws, nullptr, (float*)d_out, 1.0f);
    (void)in_sizes; (void)n_in; (void)out_size; (void)ws_size;
}
